// CapsuleLayer_71047349010982
// MI455X (gfx1250) — compile-verified, matrix operand order checked by reading, not run on hardware
//
#include <hip/hip_runtime.h>
#include <hip/hip_bf16.h>
#include <math.h>

// Problem constants (from reference)
#define N_OUT 10
#define D_OUT 16
#define D_IN  8
#define N_IN  1152
#define ROUTINGS 3
#define B_TOTAL 128
#define EPSQ 1e-7f

typedef __attribute__((ext_vector_type(2)))  float    v2f;
typedef __attribute__((ext_vector_type(8)))  float    v8f;
typedef __attribute__((ext_vector_type(16))) _Float16 v16h;

#if defined(__has_builtin)
#if __has_builtin(__builtin_amdgcn_wmma_f32_16x16x4_f32)
#define HAVE_WMMA_F32X4 1
#endif
#endif

// -----------------------------------------------------------------------------
// Kernel A: u_hat producer.
//   u_hat[b,j,i,d] = sum_k W[j,i,d,k] * x[b,i,k]
// Grid: (N_IN, N_OUT). Block: 32 * (bc/16) threads; wave w owns a 16-batch tile.
// Per wave+k-step this is a 16(M=batch) x 16(N=d) x 4(K) f32 WMMA; K=8 -> 2 WMMAs.
// ws layout (chunk-local): u_hat[b_local][j][i][d]
// -----------------------------------------------------------------------------
__global__ void __launch_bounds__(256)
capsule_uhat_wmma(const float* __restrict__ x, const float* __restrict__ Wp,
                  float* __restrict__ uh, int b0)
{
    const int i    = blockIdx.x;          // input capsule
    const int j    = blockIdx.y;          // output capsule
    const int wave = threadIdx.x >> 5;    // 16-batch tile within chunk
    const int lane = threadIdx.x & 31;
    const int l16  = lane & 15;
    const int half = lane >> 4;

    const int  d   = l16;                         // B/C column for this lane
    const int  brow = b0 + 16 * wave + l16;       // A row (batch) for this lane

    // ---- A operand: x[brow, i, k], ISA 32-bit A 16x4 layout ----
    // lanes 0-15 carry K={0,1}, lanes 16-31 carry K={2,3} (per k-step).
    const float* xr = x + ((size_t)brow * N_IN + i) * D_IN;
    v2f a0 = *(const v2f*)(xr + 2 * half);        // K = 2*half + {0,1}
    v2f a1 = *(const v2f*)(xr + 4 + 2 * half);    // K = 4 + 2*half + {0,1}

    // ---- B operand: B[k][d] = W[j,i,d,k]; k is fastest in memory -> b64 loads.
    // Same K split as A: lanes 0-15 rows {0,1}, lanes 16-31 rows {2,3}.
    const float* wr = Wp + (((size_t)j * N_IN + i) * D_OUT + d) * D_IN;
    v2f bv0 = *(const v2f*)(wr + 2 * half);
    v2f bv1 = *(const v2f*)(wr + 4 + 2 * half);

    v8f c8 = {};
#if defined(HAVE_WMMA_F32X4)
    // D = A*B + C, full f32 precision, K accumulated in two x4 steps
    c8 = __builtin_amdgcn_wmma_f32_16x16x4_f32(false, a0, false, bv0,
                                               (short)0, c8, false, false);
    c8 = __builtin_amdgcn_wmma_f32_16x16x4_f32(false, a1, false, bv1,
                                               (short)0, c8, false, false);
#else
    // Fallback: zero-padded K inside a single f16 16x16x32 WMMA (f32 accum).
    // 16-bit A layout: lanes 0-15 hold K=0..7 in elements 0..7; rest zero.
    v16h af = {};
    v16h bf = {};
    if (half == 0) {
        af[0] = (_Float16)a0[0]; af[1] = (_Float16)a0[1];      // K0,K1
        af[4] = (_Float16)a1[0]; af[5] = (_Float16)a1[1];      // K4,K5
        bf[0] = (_Float16)bv0[0]; bf[1] = (_Float16)bv0[1];
        bf[4] = (_Float16)bv1[0]; bf[5] = (_Float16)bv1[1];
        // refill K2,K3,K6,K7 from memory (this lane's row holds all 8 k's)
        af[2] = (_Float16)xr[2]; af[3] = (_Float16)xr[3];
        af[6] = (_Float16)xr[6]; af[7] = (_Float16)xr[7];
        bf[2] = (_Float16)wr[2]; bf[3] = (_Float16)wr[3];
        bf[6] = (_Float16)wr[6]; bf[7] = (_Float16)wr[7];
    }
    c8 = __builtin_amdgcn_wmma_f32_16x16x32_f16(false, af, false, bf,
                                                (short)0, c8, false, false);
#endif

    // ---- store D: VGPR r -> row M = r + 8*half, col N = l16 (=d) ----
    const size_t per_b  = (size_t)N_OUT * N_IN * D_OUT;
    const size_t jidoff = ((size_t)j * N_IN + i) * D_OUT + d;
#pragma unroll
    for (int r = 0; r < 8; ++r) {
        int bl = 16 * wave + r + 8 * half;        // chunk-local batch index
        uh[(size_t)bl * per_b + jidoff] = c8[r];
    }
}

// -----------------------------------------------------------------------------
// Kernel B: dynamic routing, one block per batch element (chunk-local).
// LDS: c[10][1152] logits + per-column softmax stats (~56 KB).
// u_hat streamed from the (L2-resident) workspace.
// -----------------------------------------------------------------------------
__global__ void __launch_bounds__(256)
capsule_routing(const float* __restrict__ uh_all, float* __restrict__ out, int b0)
{
    __shared__ float c_sh[N_OUT * N_IN];   // routing logits
    __shared__ float mx_sh[N_IN];          // softmax max per input capsule
    __shared__ float invd_sh[N_IN];        // softmax 1/denominator
    __shared__ float s_sh[N_OUT * D_OUT];
    __shared__ float v_sh[N_OUT * D_OUT];

    const int tid = threadIdx.x;
    const int bl  = blockIdx.x;                              // chunk-local batch
    const float* uh = uh_all + (size_t)bl * N_OUT * N_IN * D_OUT;

    for (int idx = tid; idx < N_OUT * N_IN; idx += 256) c_sh[idx] = 0.0f;
    __syncthreads();

    for (int r = 0; r < ROUTINGS; ++r) {
        // softmax over output capsules j, per input capsule i
        for (int i = tid; i < N_IN; i += 256) {
            float m = -3.4e38f;
#pragma unroll
            for (int j = 0; j < N_OUT; ++j) m = fmaxf(m, c_sh[j * N_IN + i]);
            float den = 0.0f;
#pragma unroll
            for (int j = 0; j < N_OUT; ++j) den += __expf(c_sh[j * N_IN + i] - m);
            mx_sh[i]   = m;
            invd_sh[i] = 1.0f / den;
        }
        __syncthreads();

        // s[j,d] = sum_i p[j,i] * u_hat[j,i,d]
        if (tid < N_OUT * D_OUT) {
            const int j = tid >> 4, d = tid & 15;
            const float* up = uh + (size_t)j * N_IN * D_OUT + d;
            float acc = 0.0f;
            for (int i = 0; i < N_IN; ++i) {
                float p = __expf(c_sh[j * N_IN + i] - mx_sh[i]) * invd_sh[i];
                acc = fmaf(p, up[(size_t)i * D_OUT], acc);
            }
            s_sh[tid] = acc;
        }
        __syncthreads();

        // squash
        if (tid < N_OUT * D_OUT) {
            const int j = tid >> 4;
            float sq = 0.0f;
#pragma unroll
            for (int d = 0; d < D_OUT; ++d) {
                float sv = s_sh[(j << 4) + d];
                sq = fmaf(sv, sv, sq);
            }
            float scale = (sq / (1.0f + sq)) * __frsqrt_rn(sq + EPSQ);
            v_sh[tid] = s_sh[tid] * scale;
        }
        __syncthreads();

        // agreement update c[j,i] += v[j,:] . u_hat[j,i,:]
        if (r < ROUTINGS - 1) {
            for (int idx = tid; idx < N_OUT * N_IN; idx += 256) {
                const int j = idx / N_IN;
                const float* up = uh + (size_t)idx * D_OUT;
                float dot = 0.0f;
#pragma unroll
                for (int d = 0; d < D_OUT; ++d)
                    dot = fmaf(v_sh[(j << 4) + d], up[d], dot);
                c_sh[idx] += dot;
            }
            __syncthreads();
        }
    }

    if (tid < N_OUT * D_OUT)
        out[(size_t)(b0 + bl) * (N_OUT * D_OUT) + tid] = v_sh[tid];
}

// -----------------------------------------------------------------------------
// Host launcher. Chunks the batch so the u_hat slab fits in ws_size.
// -----------------------------------------------------------------------------
extern "C" void kernel_launch(void* const* d_in, const int* in_sizes, int n_in,
                              void* d_out, int out_size, void* d_ws, size_t ws_size,
                              hipStream_t stream) {
    (void)in_sizes; (void)n_in; (void)out_size;
    const float* x  = (const float*)d_in[0];   // [128,1152,8]
    const float* Wp = (const float*)d_in[1];   // [10,1152,16,8]
    float* out = (float*)d_out;                // [128,10,16]
    float* ws  = (float*)d_ws;

    const size_t per_b_bytes = (size_t)N_OUT * N_IN * D_OUT * sizeof(float); // 737,280
    int bc = B_TOTAL;                          // batch chunk (multiple of 16)
    while (bc > 16 && (size_t)bc * per_b_bytes > ws_size) bc >>= 1;

    for (int b0 = 0; b0 < B_TOTAL; b0 += bc) {
        dim3 gridA(N_IN, N_OUT);
        int  threadsA = 32 * (bc / 16);
        capsule_uhat_wmma<<<gridA, threadsA, 0, stream>>>(x, Wp, ws, b0);
        capsule_routing<<<bc, 256, 0, stream>>>(ws, out, b0);
    }
}
